// KANBinaryClassifier_76699525972279
// MI455X (gfx1250) — matrix operand order repaired and checked by reading, not compiled, not run on hardware
//
#include <hip/hip_runtime.h>
#include <hip/hip_bf16.h>

// KAN classifier, MI455X (gfx1250, wave32).
// psi stage: per-wave 16-sample tile, K=4096 reduction done on the matrix pipe
// via v_wmma_f32_16x16x32_bf16 with a block-diagonal (per-q) B operand.
// Basis exps on VALU/trans pipe (v_exp_f32), co-executing with WMMA.

typedef __attribute__((ext_vector_type(16))) __bf16 v16bf;
typedef __attribute__((ext_vector_type(8)))  float  v8f;

#define NB     32768
#define ND     64
#define NQ     8
#define NM     8
#define KTOT   4096          // NQ*ND*NM
#define L2E    1.44269504088896340736f

__global__ __launch_bounds__(256) void kan_kernel(
    const float* __restrict__ x,
    const float* __restrict__ psi_c,
    const float* __restrict__ psi_w,
    const float* __restrict__ psi_lg,
    const float* __restrict__ phi_c,
    const float* __restrict__ phi_w,
    const float* __restrict__ phi_lg,
    const float* __restrict__ bias,
    float* __restrict__ out)
{
    // ---- LDS ----
    __shared__ float  sC [KTOT];        // centers
    __shared__ float  sNG[KTOT];        // -exp(log_gamma) * log2(e)
    __shared__ __bf16 sW [KTOT];        // psi weights (bf16, WMMA B operand)
    __shared__ float  sPc[NQ*NM], sPg[NQ*NM], sPw[NQ*NM];   // phi params
    __shared__ float  sS [8][16][17];   // per-wave S tiles (pad to dodge banks)

    const int tid = threadIdx.x;

    // ---- preprocess params into LDS ----
    for (int k = tid; k < KTOT; k += 256) {
        float g  = __builtin_amdgcn_exp2f(psi_lg[k] * L2E);  // exp(log_gamma)
        sC [k] = psi_c[k];
        sNG[k] = -g * L2E;
        sW [k] = (__bf16)psi_w[k];
    }
    if (tid < NQ*NM) {
        float g = __builtin_amdgcn_exp2f(phi_lg[tid] * L2E);
        sPc[tid] = phi_c[tid];
        sPg[tid] = -g * L2E;
        sPw[tid] = phi_w[tid];
    }
    __syncthreads();

    // ---- psi stage: one wave per 16-sample tile ----
    const int wave = tid >> 5;
    const int lane = tid & 31;
    const int r    = lane & 15;     // A row / C-D column index for this lane
    const int h    = lane >> 4;     // lane half: selects K sub-ranges
    const int sbase = blockIdx.x * 128 + wave * 16;

    // Preload the 32 x values this lane ever needs (pattern repeats per q):
    // xr[2t]   -> x[row, 4t + h]      (K-group h of tile t)
    // xr[2t+1] -> x[row, 4t + h + 2]  (K-group h+2 of tile t)
    const float* xrow = x + (size_t)(sbase + r) * ND;
    float xr[32];
#pragma unroll
    for (int t = 0; t < 16; ++t) {
        xr[2*t]   = xrow[4*t + h];
        xr[2*t+1] = xrow[4*t + h + 2];
    }

    v8f acc = {0.f, 0.f, 0.f, 0.f, 0.f, 0.f, 0.f, 0.f};

    for (int q = 0; q < NQ; ++q) {
        const int kq = q * 512;
        const unsigned colmask = (r == q) ? 0xFFFFFFFFu : 0u; // B nonzero only in col q
#pragma unroll
        for (int t = 0; t < 16; ++t) {
            const int kb = kq + t * 32;
            const int k0 = kb + h * 8;        // K_local = h*8 .. h*8+7
            const int k1 = kb + 16 + h * 8;   // K_local = 16+h*8 ..
            const float x0 = xr[2*t], x1 = xr[2*t + 1];

            // A operand: 16 basis values per lane, bf16
            v16bf a;
#pragma unroll
            for (int m = 0; m < NM; ++m) {
                float t0 = x0 - sC[k0 + m];
                float t1 = x1 - sC[k1 + m];
                float e0 = __builtin_amdgcn_exp2f(sNG[k0 + m] * t0 * t0);
                float e1 = __builtin_amdgcn_exp2f(sNG[k1 + m] * t1 * t1);
                a[m]     = (__bf16)e0;
                a[8 + m] = (__bf16)e1;
            }

            // B operand: rows K_local = h*8 .. h*8+7 and 16+h*8 .. 16+h*8+7 of the 32x16 weight
            // tile; only column q nonzero -> mask at dword granularity.
            const int wb = kb + h * 8;
            union { v16bf v; unsigned u[8]; } bu;
#pragma unroll
            for (int i = 0; i < 16; ++i) bu.v[i] = sW[wb + (i < 8 ? i : 8 + i)];
#pragma unroll
            for (int i = 0; i < 8; ++i) bu.u[i] &= colmask;

            acc = __builtin_amdgcn_wmma_f32_16x16x32_bf16(
                false, a, false, bu.v, (short)0, acc, false, false);
        }
    }

    // Stage S tile: C layout -> lane holds C[row = v + 8h, col = r]
#pragma unroll
    for (int v = 0; v < 8; ++v) sS[wave][v + 8*h][r] = acc[v];
    __syncthreads();

    // ---- phi stage: 128 threads, one sample each (64 exps/sample) ----
    if (tid < 128) {
        const int w2 = tid >> 4, r2 = tid & 15;
        float tot = 0.f;
#pragma unroll
        for (int q = 0; q < NQ; ++q) {
            const float s = sS[w2][r2][q];
#pragma unroll
            for (int m = 0; m < NM; ++m) {
                const int i = q * NM + m;
                float d = s - sPc[i];
                tot += sPw[i] * __builtin_amdgcn_exp2f(sPg[i] * d * d);
            }
        }
        out[blockIdx.x * 128 + tid] = tot + bias[0];
    }
}

extern "C" void kernel_launch(void* const* d_in, const int* in_sizes, int n_in,
                              void* d_out, int out_size, void* d_ws, size_t ws_size,
                              hipStream_t stream) {
    (void)in_sizes; (void)n_in; (void)d_ws; (void)ws_size; (void)out_size;
    const float* x      = (const float*)d_in[0];
    const float* psi_c  = (const float*)d_in[1];
    const float* psi_w  = (const float*)d_in[2];
    const float* psi_lg = (const float*)d_in[3];
    const float* phi_c  = (const float*)d_in[4];
    const float* phi_w  = (const float*)d_in[5];
    const float* phi_lg = (const float*)d_in[6];
    const float* bias   = (const float*)d_in[7];
    float* out = (float*)d_out;

    dim3 grid(NB / 128), block(256);
    kan_kernel<<<grid, block, 0, stream>>>(x, psi_c, psi_w, psi_lg,
                                           phi_c, phi_w, phi_lg, bias, out);
}
